// SemanticAlignmentModule_11072425689251
// MI455X (gfx1250) — matrix operand order repaired and checked by reading, not compiled, not run on hardware
//
#include <hip/hip_runtime.h>
#include <hip/hip_bf16.h>

typedef __attribute__((ext_vector_type(8)))  __bf16 v8bf;
typedef __attribute__((ext_vector_type(16))) __bf16 v16bf;
typedef __attribute__((ext_vector_type(8)))  float  v8f;

#define C_DIM   512
#define T_TOK   77
#define NTOK    4096
#define BN      32768           /* B * N = 8 * 4096 */
#define BT      616             /* B * T = 8 * 77   */
#define BT_PAD  640
#define HID     2048
#define NEG_INF (-1e30f)

// ---------------- helpers ----------------

__device__ inline __bf16 f2bf(float x) {
  union { float f; unsigned u; } a; a.f = x;
  unsigned r = a.u + 0x7FFFu + ((a.u >> 16) & 1u);   // round-to-nearest-even
  union { unsigned short s; __bf16 b; } o; o.s = (unsigned short)(r >> 16);
  return o.b;
}

__device__ inline float wsum(float v) {
#pragma unroll
  for (int o = 16; o > 0; o >>= 1) v += __shfl_xor(v, o, 32);
  return v;
}
__device__ inline float wmaxr(float v) {
#pragma unroll
  for (int o = 16; o > 0; o >>= 1) v = fmaxf(v, __shfl_xor(v, o, 32));
  return v;
}

// ---------------- weight pack:  W[K,N] fp32 -> WMMA-B-layout bf16 ----------------
// out element o = ((kt*Nt16 + nt)*32 + lane)*16 + j  holds  W[kt*32 + (lane>>4)*8 + (j&7) + (j>>3)*16, nt*16 + (lane&15)]
__global__ void k_pack_w(const float* __restrict__ W, __bf16* __restrict__ out, int K, int N) {
  long long o = (long long)blockIdx.x * 256 + threadIdx.x;
  if (o >= (long long)K * N) return;
  int j    = (int)(o & 15);
  int lane = (int)((o >> 4) & 31);
  long long blk = o >> 9;
  int Nt16 = N >> 4;
  int nt = (int)(blk % Nt16);
  int kt = (int)(blk / Nt16);
  int kk = kt * 32 + ((lane >> 4) << 3) + (j & 7) + ((j >> 3) << 4);
  int nn = nt * 16 + (lane & 15);
  out[o] = f2bf(W[(long long)kk * N + nn]);
}

__global__ void k_f2bf(const float* __restrict__ in, __bf16* __restrict__ out, int n) {
  int i = blockIdx.x * 256 + threadIdx.x;
  if (i < n) out[i] = f2bf(in[i]);
}

// ---------------- LN1 (+ gate dot) : wave per row ----------------
__global__ __launch_bounds__(128) void k_ln1(const float* __restrict__ vis,
    const float* __restrict__ g, const float* __restrict__ b,
    const float* __restrict__ Wg, const float* __restrict__ bg,
    float* __restrict__ x, __bf16* __restrict__ xbf, float* __restrict__ gate) {
  int widx = threadIdx.x >> 5, lane = threadIdx.x & 31;
  int row = blockIdx.x * 4 + widx;
  const float* r = vis + (size_t)row * C_DIM;
  float vals[16]; float s = 0.f;
#pragma unroll
  for (int j = 0; j < 16; ++j) { vals[j] = r[lane + 32 * j]; s += vals[j]; }
  s = wsum(s);
  float mu = s * (1.f / C_DIM);
  float vv = 0.f;
#pragma unroll
  for (int j = 0; j < 16; ++j) { float d = vals[j] - mu; vv += d * d; }
  vv = wsum(vv);
  float rstd = rsqrtf(vv * (1.f / C_DIM) + 1e-5f);
  float gd = 0.f;
#pragma unroll
  for (int j = 0; j < 16; ++j) {
    int c = lane + 32 * j;
    float y = (vals[j] - mu) * rstd * g[c] + b[c];
    x[(size_t)row * C_DIM + c] = y;
    xbf[(size_t)row * C_DIM + c] = f2bf(y);
    gd += y * Wg[c];
  }
  gd = wsum(gd);
  if (lane == 0) gate[row] = 1.f / (1.f + __expf(-(gd + bg[0])));
}

// ---------------- pad mask: wave per (b,t) row ----------------
__global__ __launch_bounds__(128) void k_padmask(const float* __restrict__ text,
                                                 float* __restrict__ pad) {
  int widx = threadIdx.x >> 5, lane = threadIdx.x & 31;
  int row = blockIdx.x * 4 + widx;      // 616 rows exactly
  const float* r = text + (size_t)row * C_DIM;
  float s = 0.f;
#pragma unroll
  for (int j = 0; j < 16; ++j) s += fabsf(r[lane + 32 * j]);
  s = wsum(s);
  if (lane == 0) pad[row] = (s <= 1e-6f) ? 1.f : 0.f;
}

// ---------------- l2-normalize rows (in place, fp32) ----------------
__global__ __launch_bounds__(128) void k_l2norm(float* __restrict__ buf) {
  int widx = threadIdx.x >> 5, lane = threadIdx.x & 31;
  int row = blockIdx.x * 4 + widx;
  float* p = buf + (size_t)row * C_DIM;
  float vals[16]; float ss = 0.f;
#pragma unroll
  for (int j = 0; j < 16; ++j) { vals[j] = p[lane + 32 * j]; ss += vals[j] * vals[j]; }
  ss = wsum(ss);
  float inv = 1.f / fmaxf(sqrtf(ss), 1e-6f);
#pragma unroll
  for (int j = 0; j < 16; ++j) p[lane + 32 * j] = vals[j] * inv;
}

// ---------------- WMMA bf16 GEMM:  C[M,N] = A[M,K] @ Bpk + bias ----------------
// mode 0: fp32 out;  mode 1: exact GELU, bf16 out
__device__ inline v16bf load_a16x32(const __bf16* A, int lda, int row, int kb) {
  const __bf16* p = A + (size_t)row * lda + kb;
  v8bf lo = *(const v8bf*)p;
  v8bf hi = *(const v8bf*)(p + 16);
  return __builtin_shufflevector(lo, hi, 0,1,2,3,4,5,6,7,8,9,10,11,12,13,14,15);
}

__global__ __launch_bounds__(256) void k_gemm(const __bf16* __restrict__ A,
    const __bf16* __restrict__ Bpk, const float* __restrict__ bias,
    float* __restrict__ Cf, __bf16* __restrict__ Cbf,
    int M, int Mt32, int N, int K, int mode) {
  int wid = (blockIdx.x * 256 + threadIdx.x) >> 5;
  int lane = threadIdx.x & 31;
  int Nt32 = N >> 5, Nt16 = N >> 4;
  if (wid >= Mt32 * Nt32) return;
  int mt = wid / Nt32, nt = wid % Nt32;
  int m0 = mt << 5, n0 = nt << 5;
  int cl = lane & 15, kh = lane >> 4;
  int r0 = m0 + cl;       if (r0 > M - 1) r0 = M - 1;   // clamp ragged M
  int r1 = m0 + 16 + cl;  if (r1 > M - 1) r1 = M - 1;
  v8f c00 = {}, c01 = {}, c10 = {}, c11 = {};
  for (int k0 = 0; k0 < K; k0 += 32) {
    int kb = k0 + (kh << 3);
    v16bf a0 = load_a16x32(A, K, r0, kb);
    v16bf a1 = load_a16x32(A, K, r1, kb);
    const __bf16* bp = Bpk + (((size_t)(k0 >> 5) * Nt16 + (n0 >> 4)) * 32 + lane) * 16;
    v16bf b0 = *(const v16bf*)bp;
    v16bf b1 = *(const v16bf*)(bp + 512);
    c00 = __builtin_amdgcn_wmma_f32_16x16x32_bf16(false, a0, false, b0, (short)0, c00, false, false);
    c01 = __builtin_amdgcn_wmma_f32_16x16x32_bf16(false, a0, false, b1, (short)0, c01, false, false);
    c10 = __builtin_amdgcn_wmma_f32_16x16x32_bf16(false, a1, false, b0, (short)0, c10, false, false);
    c11 = __builtin_amdgcn_wmma_f32_16x16x32_bf16(false, a1, false, b1, (short)0, c11, false, false);
  }
  float bs0 = bias[n0 + cl], bs1 = bias[n0 + 16 + cl];
  int rm = kh << 3;
#pragma unroll
  for (int i = 0; i < 8; ++i) {
    int ra = m0 + rm + i;
    int rb = m0 + 16 + rm + i;
    float v00 = c00[i] + bs0, v01 = c01[i] + bs1;
    float v10 = c10[i] + bs0, v11 = c11[i] + bs1;
    if (mode == 1) {
      v00 = 0.5f * v00 * (1.f + erff(v00 * 0.70710678f));
      v01 = 0.5f * v01 * (1.f + erff(v01 * 0.70710678f));
      v10 = 0.5f * v10 * (1.f + erff(v10 * 0.70710678f));
      v11 = 0.5f * v11 * (1.f + erff(v11 * 0.70710678f));
      if (ra < M) { Cbf[(size_t)ra * N + n0 + cl] = f2bf(v00); Cbf[(size_t)ra * N + n0 + 16 + cl] = f2bf(v01); }
      if (rb < M) { Cbf[(size_t)rb * N + n0 + cl] = f2bf(v10); Cbf[(size_t)rb * N + n0 + 16 + cl] = f2bf(v11); }
    } else {
      if (ra < M) { Cf[(size_t)ra * N + n0 + cl] = v00; Cf[(size_t)ra * N + n0 + 16 + cl] = v01; }
      if (rb < M) { Cf[(size_t)rb * N + n0 + cl] = v10; Cf[(size_t)rb * N + n0 + 16 + cl] = v11; }
    }
  }
}

// ---------------- attention: wave per (b,n); top-3, softmax, confidence, attn@V ----------------
__global__ __launch_bounds__(128) void k_attn(const float* __restrict__ q,
    const float* __restrict__ k, const float* __restrict__ v,
    const float* __restrict__ pad, const float* __restrict__ ls,
    __bf16* __restrict__ aligned, float* __restrict__ bypass) {
  __shared__ float qs[4][C_DIM];
  __shared__ float attn_s[4][96];
  int widx = threadIdx.x >> 5, lane = threadIdx.x & 31;
  int wg = blockIdx.x * 4 + widx;          // (b, n) flat
  int b = wg >> 12;                        // / NTOK
  const float* qrow = q + (size_t)wg * C_DIM;
  for (int i = lane; i < C_DIM; i += 32) qs[widx][i] = qrow[i];
  __syncthreads();
  float scale = __expf(fminf(fmaxf(ls[0], -2.f), 2.f)) * 0.125f;  // /sqrt(64)
  float padv[3]; float pc = 0.f;
#pragma unroll
  for (int j = 0; j < 3; ++j) {
    int t = lane + 32 * j;
    padv[j] = (t < T_TOK) ? pad[b * T_TOK + t] : 1.f;
    if (t < T_TOK) pc += padv[j];
  }
  pc = wsum(pc);
  bool all_pad = (pc >= (float)T_TOK - 0.5f);
  float conf_sum = 0.f;
  for (int h = 0; h < 8; ++h) {
    float s[3];
#pragma unroll
    for (int j = 0; j < 3; ++j) {
      int t = lane + 32 * j;
      float sv = NEG_INF;
      if (t < T_TOK && padv[j] < 0.5f) {
        const float4* kp = (const float4*)(k + ((size_t)(b * T_TOK + t)) * C_DIM + h * 64);
        const float4* qp = (const float4*)(&qs[widx][h * 64]);
        float acc = 0.f;
#pragma unroll
        for (int d4 = 0; d4 < 16; ++d4) {
          float4 kv = kp[d4]; float4 qv = qp[d4];
          acc += kv.x * qv.x + kv.y * qv.y + kv.z * qv.z + kv.w * qv.w;
        }
        sv = acc * scale;
      }
      s[j] = sv;
    }
    // top-3 threshold with multiplicity: 3 rounds of wave-max + single-instance removal
    float t0 = s[0], t1 = s[1], t2 = s[2];
    float thr = NEG_INF, m1 = NEG_INF;
    for (int r = 0; r < 3; ++r) {
      float lm = fmaxf(t0, fmaxf(t1, t2));
      float gm = wmaxr(lm);
      if (r == 0) m1 = gm;
      thr = gm;
      unsigned long long mb = __ballot(lm == gm);
      int leader = __ffsll(mb) - 1;
      if (lane == leader) {
        if (t0 == gm) t0 = -__builtin_inff();
        else if (t1 == gm) t1 = -__builtin_inff();
        else t2 = -__builtin_inff();
      }
    }
    float e[3], sl = 0.f, cl_ = 0.f, ml = 0.f;
#pragma unroll
    for (int j = 0; j < 3; ++j) {
      int t = lane + 32 * j;
      float ev = 0.f;
      if (t < T_TOK && !all_pad && s[j] >= thr) ev = __expf(s[j] - m1);
      e[j] = ev; sl += ev; if (ev > 0.f) cl_ += 1.f; ml = fmaxf(ml, ev);
    }
    float sum = wsum(sl);
    float cnt = wsum(cl_);
    float emax = wmaxr(ml);
    float inv = 1.f / fmaxf(sum, 1e-6f);
    float maxp = emax * inv;
    float teff = fmaxf(cnt, 2.f);
    float entl = 0.f;
#pragma unroll
    for (int j = 0; j < 3; ++j) {
      int t = lane + 32 * j;
      if (t < T_TOK) {
        float p = fmaxf(e[j] * inv, 1e-8f);
        entl += -p * __logf(p);
      }
    }
    float ent = fmaxf(wsum(entl) / __logf(teff), 0.f);
    conf_sum += fminf(fmaxf(maxp * (1.f - ent), 0.f), 1.f);
    // broadcast attention row via LDS, then attn @ V for this head
#pragma unroll
    for (int j = 0; j < 3; ++j) {
      int t = lane + 32 * j;
      if (t < 96) attn_s[widx][t] = (t < T_TOK) ? e[j] * inv : 0.f;
    }
    __syncthreads();
#pragma unroll
    for (int jj = 0; jj < 2; ++jj) {
      int d = lane + 32 * jj;
      float acc = 0.f;
      for (int t = 0; t < T_TOK; ++t)
        acc += attn_s[widx][t] * v[((size_t)(b * T_TOK + t)) * C_DIM + h * 64 + d];
      aligned[(size_t)wg * C_DIM + h * 64 + d] = f2bf(acc);
    }
    __syncthreads();
  }
  if (lane == 0) bypass[wg] = (conf_sum * 0.125f >= 0.35f) ? 1.f : 0.f;
}

// ---------------- gated residual + LN2 : wave per row ----------------
__global__ __launch_bounds__(128) void k_combine_ln2(const float* __restrict__ x,
    const float* __restrict__ att, const float* __restrict__ gate,
    const float* __restrict__ byp, const float* __restrict__ alpha,
    const float* __restrict__ g, const float* __restrict__ b,
    float* __restrict__ yln, __bf16* __restrict__ ylnbf) {
  int widx = threadIdx.x >> 5, lane = threadIdx.x & 31;
  int row = blockIdx.x * 4 + widx;
  float ga = alpha[0] * gate[row] * byp[row];
  const float* xr = x + (size_t)row * C_DIM;
  const float* ar = att + (size_t)row * C_DIM;
  float vals[16]; float s = 0.f;
#pragma unroll
  for (int j = 0; j < 16; ++j) {
    int c = lane + 32 * j;
    float y = xr[c] + ga * ar[c];
    vals[j] = y; s += y;
  }
  s = wsum(s); float mu = s * (1.f / C_DIM);
  float vv = 0.f;
#pragma unroll
  for (int j = 0; j < 16; ++j) { float d = vals[j] - mu; vv += d * d; }
  vv = wsum(vv);
  float rstd = rsqrtf(vv * (1.f / C_DIM) + 1e-5f);
#pragma unroll
  for (int j = 0; j < 16; ++j) {
    int c = lane + 32 * j;
    float o = (vals[j] - mu) * rstd * g[c] + b[c];
    yln[(size_t)row * C_DIM + c] = o;
    ylnbf[(size_t)row * C_DIM + c] = f2bf(o);
  }
}

__global__ void k_add(const float* __restrict__ a, const float* __restrict__ b,
                      float* __restrict__ o, int n) {
  int i = blockIdx.x * 256 + threadIdx.x;
  if (i < n) o[i] = a[i] + b[i];
}

// ---------------- host ----------------

extern "C" void kernel_launch(void* const* d_in, const int* in_sizes, int n_in,
                              void* d_out, int out_size, void* d_ws, size_t ws_size,
                              hipStream_t stream) {
  (void)in_sizes; (void)n_in; (void)out_size; (void)ws_size;
  const float* vis  = (const float*)d_in[0];
  const float* text = (const float*)d_in[1];
  const float* ln1g = (const float*)d_in[2];
  const float* ln1b = (const float*)d_in[3];
  const float* ln2g = (const float*)d_in[4];
  const float* ln2b = (const float*)d_in[5];
  const float* Wq = (const float*)d_in[6];   const float* bq = (const float*)d_in[7];
  const float* Wk = (const float*)d_in[8];   const float* bk = (const float*)d_in[9];
  const float* Wv = (const float*)d_in[10];  const float* bv = (const float*)d_in[11];
  const float* Wo = (const float*)d_in[12];  const float* bo = (const float*)d_in[13];
  const float* Wg = (const float*)d_in[14];  const float* bg = (const float*)d_in[15];
  const float* W1 = (const float*)d_in[16];  const float* b1 = (const float*)d_in[17];
  const float* W2 = (const float*)d_in[18];  const float* b2 = (const float*)d_in[19];
  const float* ls = (const float*)d_in[20];
  const float* alpha = (const float*)d_in[21];
  float* out = (float*)d_out;

  char* ws = (char*)d_ws;
  size_t off = 0;
  auto alloc = [&](size_t bytes) -> void* {
    void* p = ws + off; off = (off + bytes + 255) & ~(size_t)255; return p;
  };
  const size_t BNC = (size_t)BN * C_DIM;          // 16,777,216
  float*  x_f     = (float*) alloc(BNC * 4);
  __bf16* x_bf    = (__bf16*)alloc(BNC * 2);
  float*  gate    = (float*) alloc((size_t)BN * 4);
  __bf16* text_bf = (__bf16*)alloc((size_t)BT * C_DIM * 2);
  __bf16* Wq_pk   = (__bf16*)alloc((size_t)C_DIM * C_DIM * 2);
  __bf16* Wk_pk   = (__bf16*)alloc((size_t)C_DIM * C_DIM * 2);
  __bf16* Wv_pk   = (__bf16*)alloc((size_t)C_DIM * C_DIM * 2);
  __bf16* Wo_pk   = (__bf16*)alloc((size_t)C_DIM * C_DIM * 2);
  __bf16* W1_pk   = (__bf16*)alloc((size_t)C_DIM * HID * 2);
  __bf16* W2_pk   = (__bf16*)alloc((size_t)HID * C_DIM * 2);
  float*  q_f     = (float*) alloc(BNC * 4);
  float*  k_f     = (float*) alloc((size_t)BT_PAD * C_DIM * 4);
  float*  v_f     = (float*) alloc((size_t)BT_PAD * C_DIM * 4);
  float*  pad     = (float*) alloc((size_t)BT * 4);
  __bf16* alg_bf  = (__bf16*)alloc(BNC * 2);
  float*  byp     = (float*) alloc((size_t)BN * 4);
  float*  att_out = (float*) alloc(BNC * 4);
  float*  yln     = (float*) alloc(BNC * 4);
  __bf16* yln_bf  = (__bf16*)alloc(BNC * 2);
  __bf16* h1_bf   = (__bf16*)alloc((size_t)BN * HID * 2);
  float*  h2_f    = (float*) alloc(BNC * 4);

  // 1. weight packing (WMMA B layout, bf16)
  k_pack_w<<<(C_DIM * C_DIM + 255) / 256, 256, 0, stream>>>(Wq, Wq_pk, C_DIM, C_DIM);
  k_pack_w<<<(C_DIM * C_DIM + 255) / 256, 256, 0, stream>>>(Wk, Wk_pk, C_DIM, C_DIM);
  k_pack_w<<<(C_DIM * C_DIM + 255) / 256, 256, 0, stream>>>(Wv, Wv_pk, C_DIM, C_DIM);
  k_pack_w<<<(C_DIM * C_DIM + 255) / 256, 256, 0, stream>>>(Wo, Wo_pk, C_DIM, C_DIM);
  k_pack_w<<<(C_DIM * HID   + 255) / 256, 256, 0, stream>>>(W1, W1_pk, C_DIM, HID);
  k_pack_w<<<(HID   * C_DIM + 255) / 256, 256, 0, stream>>>(W2, W2_pk, HID, C_DIM);
  // 2. text -> bf16, pad mask
  k_f2bf<<<(BT * C_DIM + 255) / 256, 256, 0, stream>>>(text, text_bf, BT * C_DIM);
  k_padmask<<<BT / 4, 128, 0, stream>>>(text, pad);
  // 3. LN1 + gate
  k_ln1<<<BN / 4, 128, 0, stream>>>(vis, ln1g, ln1b, Wg, bg, x_f, x_bf, gate);
  // 4. Q/K/V projections (WMMA)
  k_gemm<<<(1024 * 16 + 7) / 8, 256, 0, stream>>>(x_bf, Wq_pk, bq, q_f, nullptr, BN, 1024, C_DIM, C_DIM, 0);
  k_gemm<<<(20 * 16 + 7) / 8, 256, 0, stream>>>(text_bf, Wk_pk, bk, k_f, nullptr, BT, 20, C_DIM, C_DIM, 0);
  k_gemm<<<(20 * 16 + 7) / 8, 256, 0, stream>>>(text_bf, Wv_pk, bv, v_f, nullptr, BT, 20, C_DIM, C_DIM, 0);
  // 5. l2-normalize q, k
  k_l2norm<<<BN / 4, 128, 0, stream>>>(q_f);
  k_l2norm<<<BT / 4, 128, 0, stream>>>(k_f);
  // 6. sparse top-3 attention + confidence
  k_attn<<<BN / 4, 128, 0, stream>>>(q_f, k_f, v_f, pad, ls, alg_bf, byp);
  // 7. output projection (WMMA)
  k_gemm<<<(1024 * 16 + 7) / 8, 256, 0, stream>>>(alg_bf, Wo_pk, bo, att_out, nullptr, BN, 1024, C_DIM, C_DIM, 0);
  // 8. gated residual + LN2
  k_combine_ln2<<<BN / 4, 128, 0, stream>>>(x_f, att_out, gate, byp, alpha, ln2g, ln2b, yln, yln_bf);
  // 9. MLP (WMMA): GELU fused into first GEMM
  k_gemm<<<(1024 * 64 + 7) / 8, 256, 0, stream>>>(yln_bf, W1_pk, b1, nullptr, h1_bf, BN, 1024, HID, C_DIM, 1);
  k_gemm<<<(1024 * 16 + 7) / 8, 256, 0, stream>>>(h1_bf, W2_pk, b2, h2_f, nullptr, BN, 1024, C_DIM, HID, 0);
  // 10. final residual
  k_add<<<(int)((BNC + 255) / 256), 256, 0, stream>>>(yln, h2_f, out, (int)BNC);
}
